// VQ_75514114998355
// MI455X (gfx1250) — matrix operand order repaired and checked by reading, not compiled, not run on hardware
//
#include <hip/hip_runtime.h>
#include <hip/hip_bf16.h>

// ---------------------------------------------------------------------------
// VQ codebook quantization for MI455X (gfx1250, wave32, WMMA bf16 path,
// async global->LDS tile staging).
//   x:        [32, 4096, 256] f32   (131072 vectors, dim 256)
//   embedding:[1024, 256]     f32
// Outputs (flat f32, concatenated):
//   values   [33554432] = embedding[argmin]
//   indexes  [131072]   (stored as float)
//   loss     [1]        = 1.2 * mean(dist_min)   (dist_min == ||x - v||^2)
//   usages   [1024]
// ---------------------------------------------------------------------------

typedef __attribute__((ext_vector_type(16))) __bf16 v16bf;
typedef __attribute__((ext_vector_type(8)))  float  v8f;

#define DIMS      256
#define CODEBOOK  1024
#define NVEC      131072            // 32 * 4096
#define NTILES    64                // CODEBOOK / 16
#define LDS_PITCH 264               // 256 halves + 8 pad (bank spread)

__device__ __forceinline__ unsigned int pack_bf2(float lo, float hi) {
  // round-to-nearest-even f32 -> bf16, pack two into one dword
  unsigned int a = __float_as_uint(lo);
  unsigned int b = __float_as_uint(hi);
  a = (a + 0x7FFFu + ((a >> 16) & 1u)) >> 16;
  b = (b + 0x7FFFu + ((b >> 16) & 1u)) & 0xFFFF0000u;
  return a | b;
}

// ---------------- init: e2, bf16 codebook, zero outputs --------------------
__global__ __launch_bounds__(256) void vq_init_kernel(
    const float* __restrict__ emb, float* __restrict__ e2g,
    unsigned short* __restrict__ emb_bf,
    float* __restrict__ out_loss, float* __restrict__ out_usage) {
  int k = blockIdx.x * 256 + threadIdx.x;
  if (k < CODEBOOK) {
    const float4* ep = (const float4*)(emb + (size_t)k * DIMS);
    uint4* op = (uint4*)(emb_bf + (size_t)k * DIMS);   // 32 uint4 per row
    float s = 0.f;
#pragma unroll 8
    for (int j = 0; j < DIMS / 8; ++j) {
      float4 v0 = ep[2 * j];
      float4 v1 = ep[2 * j + 1];
      s += v0.x * v0.x + v0.y * v0.y + v0.z * v0.z + v0.w * v0.w;
      s += v1.x * v1.x + v1.y * v1.y + v1.z * v1.z + v1.w * v1.w;
      uint4 p;
      p.x = pack_bf2(v0.x, v0.y); p.y = pack_bf2(v0.z, v0.w);
      p.z = pack_bf2(v1.x, v1.y); p.w = pack_bf2(v1.z, v1.w);
      op[j] = p;
    }
    e2g[k] = s;
    out_usage[k] = 0.f;
    if (k == 0) *out_loss = 0.f;
  }
}

// --------------------------- main argmin kernel ----------------------------
// Block = 256 threads (8 waves). Each wave owns 2 row-tiles of 16 x-rows
// (32 rows); block covers 256 rows; grid = NVEC/256 = 512 blocks.
__global__ __launch_bounds__(256) __attribute__((amdgpu_waves_per_eu(4)))
void vq_argmin_kernel(
    const float* __restrict__ x, const unsigned short* __restrict__ emb_bf,
    const float* __restrict__ e2g, int* __restrict__ idx_ws,
    float* __restrict__ out_idx, float* __restrict__ out_loss) {
  __shared__ __align__(16) unsigned short lds[2][16 * LDS_PITCH];
  __shared__ __align__(16) float e2l[CODEBOOK];     // e2 cached in LDS (4 KB)

  const int tid  = threadIdx.x;
  const int lane = tid & 31;
  const int w    = tid >> 5;
  const int ln16 = lane & 15;
  const int hi   = lane >> 4;                 // 0 | 1 (half-wave)
  const int rowbase0 = blockIdx.x * 256 + w * 32;

  // ---- stage e2 into LDS (keeps the main loop free of LOADcnt waits) -----
  {
    float4 e4 = *(const float4*)(e2g + tid * 4);
    *(float4*)(e2l + tid * 4) = e4;
  }

  // ---- Build bf16 A fragments in registers + per-row ||x||^2 -------------
  // 16-bit A 16x32 layout: per 32-K chunk, halves 0..7 = K kb+0..kb+7,
  // halves 8..15 = K kb+16..kb+23, kb = hi*8.
  union AFrag { v16bf v; unsigned int u[8]; };
  AFrag afr[2][8];
  float x2tot[2];
#pragma unroll
  for (int t = 0; t < 2; ++t) {
    const float* xr = x + (size_t)(rowbase0 + t * 16 + ln16) * DIMS;
    float s = 0.f;
#pragma unroll
    for (int c = 0; c < 8; ++c) {
      const int col = c * 32 + hi * 8;
      float4 a0 = *(const float4*)(xr + col);
      float4 a1 = *(const float4*)(xr + col + 4);
      float4 a2 = *(const float4*)(xr + col + 16);
      float4 a3 = *(const float4*)(xr + col + 20);
      s += a0.x * a0.x + a0.y * a0.y + a0.z * a0.z + a0.w * a0.w;
      s += a1.x * a1.x + a1.y * a1.y + a1.z * a1.z + a1.w * a1.w;
      s += a2.x * a2.x + a2.y * a2.y + a2.z * a2.z + a2.w * a2.w;
      s += a3.x * a3.x + a3.y * a3.y + a3.z * a3.z + a3.w * a3.w;
      afr[t][c].u[0] = pack_bf2(a0.x, a0.y);
      afr[t][c].u[1] = pack_bf2(a0.z, a0.w);
      afr[t][c].u[2] = pack_bf2(a1.x, a1.y);
      afr[t][c].u[3] = pack_bf2(a1.z, a1.w);
      afr[t][c].u[4] = pack_bf2(a2.x, a2.y);
      afr[t][c].u[5] = pack_bf2(a2.z, a2.w);
      afr[t][c].u[6] = pack_bf2(a3.x, a3.y);
      afr[t][c].u[7] = pack_bf2(a3.z, a3.w);
    }
    x2tot[t] = s + __shfl_xor(s, 16);         // lanes l, l^16 cover one row
  }

  // ---- Async global->LDS staging of bf16 embedding tiles -----------------
  // Thread t stages 32 bytes: codeword row = t>>4, 16-half segment = t&15.
  // Two async b128 ops; inst offset applies to BOTH global and LDS address.
  const int srow = tid >> 4;
  const int sseg = tid & 15;
  const unsigned lbase =
      (unsigned)(uintptr_t)(&lds[0][0]) + (unsigned)(srow * LDS_PITCH + sseg * 16) * 2u;
  const unsigned lstride = (unsigned)sizeof(unsigned short) * 16 * LDS_PITCH;
  auto issue_async = [&](int nt, int buf) {
    unsigned long long ga =
        (unsigned long long)(emb_bf + (size_t)(nt * 16 + srow) * DIMS + sseg * 16);
    unsigned la = lbase + (unsigned)buf * lstride;
    asm volatile("global_load_async_to_lds_b128 %0, %1, off"
                 :: "v"(la), "v"(ga) : "memory");
    asm volatile("global_load_async_to_lds_b128 %0, %1, off offset:16"
                 :: "v"(la), "v"(ga) : "memory");
  };

  // best tracks d' = ||e||^2 - 2 x.e  (||x||^2 is per-row constant: argmin
  // unchanged; re-added at the end for the loss term)
  float best[2][8];
  int   bidx[2][8];
#pragma unroll
  for (int t = 0; t < 2; ++t)
#pragma unroll
    for (int r = 0; r < 8; ++r) { best[t][r] = 3.4e38f; bidx[t][r] = 0; }

  issue_async(0, 0);
  for (int nt = 0; nt < NTILES; ++nt) {
    asm volatile("s_wait_asynccnt 0x0" ::: "memory");  // this wave's tile done
    __syncthreads();                                   // all waves' tiles done
    if (nt + 1 < NTILES) issue_async(nt + 1, (nt + 1) & 1);  // overlaps WMMAs

    const float e2v = e2l[nt * 16 + ln16];    // LDS read: stays in DScnt flow
    v8f acc0 = {0.f, 0.f, 0.f, 0.f, 0.f, 0.f, 0.f, 0.f};
    v8f acc1 = {0.f, 0.f, 0.f, 0.f, 0.f, 0.f, 0.f, 0.f};
    // B 32x16 layout: lane holds col n=ln16; halves 0..7 = K kb..kb+7, 8..15 = K kb+16..kb+23, kb = hi*8
    const unsigned short* bp = &lds[nt & 1][ln16 * LDS_PITCH + hi * 8];

    // Manually double-buffered B fragments: issue ds_loads for chunk c+1
    // before the two WMMAs of chunk c so LDS latency hides under the XDL ops.
    union BFrag { v16bf v; uint4 q[2]; };
    BFrag bcur, bnxt;
    bcur.q[0] = *(const uint4*)(bp);
    bcur.q[1] = *(const uint4*)(bp + 16);
#pragma unroll
    for (int c = 0; c < 8; ++c) {
      BFrag buse = bcur;
      if (c < 7) {
        bnxt.q[0] = *(const uint4*)(bp + (c + 1) * 32);
        bnxt.q[1] = *(const uint4*)(bp + (c + 1) * 32 + 16);
        bcur = bnxt;
      }
      acc0 = __builtin_amdgcn_wmma_f32_16x16x32_bf16(
          false, afr[0][c].v, false, buse.v, (short)0, acc0, false, false);
      acc1 = __builtin_amdgcn_wmma_f32_16x16x32_bf16(
          false, afr[1][c].v, false, buse.v, (short)0, acc1, false, false);
    }
    const int nb = nt * 16 + ln16;
#pragma unroll
    for (int r = 0; r < 8; ++r) {
      float d0 = __builtin_fmaf(-2.f, acc0[r], e2v);
      if (d0 < best[0][r]) { best[0][r] = d0; bidx[0][r] = nb; }
      float d1 = __builtin_fmaf(-2.f, acc1[r], e2v);
      if (d1 < best[1][r]) { best[1][r] = d1; bidx[1][r] = nb; }
    }
  }

  // ---- Cross-lane argmin within each 16-lane group, write results --------
  float lp = 0.f;
#pragma unroll
  for (int t = 0; t < 2; ++t) {
#pragma unroll
    for (int r = 0; r < 8; ++r) {
      float d = best[t][r];
      int   bi = bidx[t][r];
#pragma unroll
      for (int off = 8; off; off >>= 1) {
        float od = __shfl_xor(d, off);
        int   oi = __shfl_xor(bi, off);
        if (od < d || (od == d && oi < bi)) { d = od; bi = oi; }
      }
      float x2m = __shfl(x2tot[t], r + hi * 8);   // ||x||^2 of row r + 8*hi
      if (ln16 == 0) {
        int m = rowbase0 + t * 16 + r + hi * 8;
        idx_ws[m]  = bi;
        out_idx[m] = (float)bi;
        lp += x2m + d;                        // dist_min == ||x - v||^2
      }
    }
  }
  lp += __shfl_xor(lp, 16);
  if (lane == 0)
    atomicAdd(out_loss, lp * (1.2f / 33554432.0f));  // 1.2 * mean over B*N*D
}

// --------------------------- gather + usages -------------------------------
// One wave per vector: copy 256-f32 codeword row, bump usage histogram.
__global__ __launch_bounds__(256) void vq_gather_kernel(
    const float* __restrict__ emb, const int* __restrict__ idx_ws,
    float* __restrict__ values, float* __restrict__ usages) {
  const int w    = threadIdx.x >> 5;
  const int lane = threadIdx.x & 31;
  const int vec  = blockIdx.x * 8 + w;
  const int idx  = idx_ws[vec];
  const float4* src = (const float4*)(emb + (size_t)idx * DIMS) + lane * 2;
  float4* dst = (float4*)(values + (size_t)vec * DIMS) + lane * 2;
  dst[0] = src[0];
  dst[1] = src[1];
  if (lane == 0) atomicAdd(&usages[idx], 1.0f);
}

// ---------------------------------------------------------------------------
extern "C" void kernel_launch(void* const* d_in, const int* in_sizes, int n_in,
                              void* d_out, int out_size, void* d_ws, size_t ws_size,
                              hipStream_t stream) {
  const float* x   = (const float*)d_in[0];   // [32,4096,256] f32
  const float* emb = (const float*)d_in[1];   // [1024,256]    f32

  float* out        = (float*)d_out;
  float* values     = out;                    // 33554432 floats
  float* out_idx    = out + 33554432;         // 131072
  float* out_loss   = out + 33685504;         // 1
  float* out_usage  = out + 33685505;         // 1024

  float*          e2g    = (float*)d_ws;                          // 1024 f32
  unsigned short* emb_bf = (unsigned short*)((char*)d_ws + 4096); // 256 Ki bf16
  int*            idx_ws = (int*)((char*)d_ws + 4096 + 524288);   // 128 Ki i32

  vq_init_kernel<<<4, 256, 0, stream>>>(emb, e2g, emb_bf, out_loss, out_usage);
  vq_argmin_kernel<<<NVEC / 256, 256, 0, stream>>>(x, emb_bf, e2g, idx_ws,
                                                   out_idx, out_loss);
  vq_gather_kernel<<<NVEC / 8, 256, 0, stream>>>(emb, idx_ws, values,
                                                 out_usage);
}
